// Attention_23759759081677
// MI455X (gfx1250) — matrix operand order repaired and checked by reading, not compiled, not run on hardware
//
#include <hip/hip_runtime.h>
#include <hip/hip_bf16.h>

#define NH 12
#define HD 64

typedef _Float16 h16 __attribute__((ext_vector_type(16)));
typedef _Float16 h8  __attribute__((ext_vector_type(8)));
typedef _Float16 h4  __attribute__((ext_vector_type(4)));
typedef float    f8  __attribute__((ext_vector_type(8)));

__device__ __forceinline__ f8 wmma_f16(h16 a, h16 b, f8 c) {
    // D = A(16x32 f16) * B(32x16 f16) + C(16x16 f32)
    return __builtin_amdgcn_wmma_f32_16x16x32_f16(
        false, a, false, b, (short)0, c, false, false);
}

// A fragment (16x32 f16): lane r=l&15 -> row r; lanes<16 hold K[0..7]+K[16..23],
// lanes>=16 hold K[8..15]+K[24..31]. Row-major source => two b128 loads.
__device__ __forceinline__ h16 load_a_frag(const _Float16* base, int row_stride, int lane) {
    int r  = lane & 15;
    int lo = (lane >> 4) << 3;              // 0 or 8
    const _Float16* p = base + r * row_stride;
    h16 out;
    *(h8*)&out       = *(const h8*)(p + lo);
    *((h8*)&out + 1) = *(const h8*)(p + 16 + lo);
    return out;
}

// B fragment (32x16 f16): lane holds column n=l&15; lanes<16 hold K[0..7]+K[16..23],
// lanes>=16 hold K[8..15]+K[24..31].
__device__ __forceinline__ h16 load_b_frag(const _Float16* nbase, int n_stride, int col0, int lane) {
    int n  = lane & 15;
    int kh = (lane >> 4) << 3;              // 0 or 8
    const _Float16* p = nbase + n * n_stride + col0 + kh;
    h16 out;
    *(h8*)&out       = *(const h8*)p;
    *((h8*)&out + 1) = *(const h8*)(p + 16);
    return out;
}

// ---------------------------------------------------------------------------
// Kernel 1: QKV projection.  X[8192x768] * qkv_w^T -> q(f16,scaled)[b,h,n,d],
// k(f16)[b,h,n,d], v^T(f16)[b,h,d,n].   Block tile 128x64, 8 waves.
// ---------------------------------------------------------------------------
__global__ __launch_bounds__(256) void qkv_kernel(
    const float* __restrict__ x, const float* __restrict__ qkv_w,
    _Float16* __restrict__ q_ws, _Float16* __restrict__ k_ws,
    _Float16* __restrict__ vt_ws) {
    __shared__ _Float16 Xs[128][40];
    __shared__ _Float16 Ws[64][40];

    const int t = threadIdx.x, wave = t >> 5, lane = t & 31;
    const int row0 = blockIdx.x * 128;
    const int col0 = blockIdx.y * 64;

    f8 acc[4];
#pragma unroll
    for (int i = 0; i < 4; i++)
#pragma unroll
        for (int e = 0; e < 8; e++) acc[i][e] = 0.f;

    for (int kb = 0; kb < 768; kb += 32) {
        // batched b128 loads (all issue before any wait), then cvt + b64 LDS stores
        float4 xv[4];
#pragma unroll
        for (int i = 0; i < 4; i++) {
            int idx = i * 256 + t, r = idx >> 3, c4 = idx & 7;
            xv[i] = *(const float4*)&x[(long)(row0 + r) * 768 + kb + c4 * 4];
        }
        float4 wv[2];
#pragma unroll
        for (int i = 0; i < 2; i++) {
            int idx = i * 256 + t, r = idx >> 3, c4 = idx & 7;
            wv[i] = *(const float4*)&qkv_w[(long)(col0 + r) * 768 + kb + c4 * 4];
        }
        if (kb + 32 < 768)
            __builtin_prefetch(&x[(long)(row0 + (t >> 1)) * 768 + kb + 32], 0, 1);
#pragma unroll
        for (int i = 0; i < 4; i++) {
            int idx = i * 256 + t, r = idx >> 3, c4 = idx & 7;
            h4 hv = {(_Float16)xv[i].x, (_Float16)xv[i].y,
                     (_Float16)xv[i].z, (_Float16)xv[i].w};
            *(h4*)&Xs[r][c4 * 4] = hv;
        }
#pragma unroll
        for (int i = 0; i < 2; i++) {
            int idx = i * 256 + t, r = idx >> 3, c4 = idx & 7;
            h4 hv = {(_Float16)wv[i].x, (_Float16)wv[i].y,
                     (_Float16)wv[i].z, (_Float16)wv[i].w};
            *(h4*)&Ws[r][c4 * 4] = hv;
        }
        __syncthreads();
        h16 afrag = load_a_frag(&Xs[wave * 16][0], 40, lane);
#pragma unroll
        for (int nt = 0; nt < 4; nt++) {
            h16 bfrag = load_b_frag(&Ws[nt * 16][0], 40, 0, lane);
            acc[nt] = wmma_f16(afrag, bfrag, acc[nt]);
        }
        __syncthreads();
    }

    const float scale = 0.125f;              // hd^-0.5 = 1/8
    const int nn = lane & 15;
    const int ho = (lane >> 4) << 3;
#pragma unroll
    for (int nt = 0; nt < 4; nt++) {
        int cbase = col0 + nt * 16;          // 16-aligned, inside one 64-block
        int which = cbase / 768;             // 0=q 1=k 2=v (uniform per tile)
        int rem   = cbase % 768;
        int h = rem >> 6, d = (rem & 63) + nn;
#pragma unroll
        for (int r = 0; r < 8; r++) {
            int m = row0 + wave * 16 + r + ho;
            int b = m >> 10, n = m & 1023;
            long bh = (long)b * NH + h;
            float v = acc[nt][r];
            if (which == 0)      q_ws[(bh * 1024 + n) * 64 + d] = (_Float16)(v * scale);
            else if (which == 1) k_ws[(bh * 1024 + n) * 64 + d] = (_Float16)v;
            else                 vt_ws[(bh * 64 + d) * 1024 + n] = (_Float16)v;
        }
    }
}

// ---------------------------------------------------------------------------
// Kernel 2: fused attention (flash-style).  One block = (b, h, 32-row q tile).
// rel bias via WMMA: relF[m][j] = q[m] . rel_pos[j]  (j = 0..62, row 63 zero),
// then relH[m][hk] = relHf[m][qt+31-hk], relW[m][wk] = relWf[m][m+31-wk].
// ---------------------------------------------------------------------------
__global__ __launch_bounds__(256) void attn_kernel(
    const _Float16* __restrict__ q_ws, const _Float16* __restrict__ k_ws,
    const _Float16* __restrict__ vt_ws,
    const float* __restrict__ rel_pos_h, const float* __restrict__ rel_pos_w,
    float* __restrict__ attn_out) {
    __shared__ union {
        float    S[32][132];                 // score chunk (per k-chunk)
        _Float16 stage[2][64][72];           // f16 rel_pos tables (phase 0 only)
    } u;
    __shared__ _Float16 P[32][136];
    __shared__ float relHf[32][68];
    __shared__ float relWf[32][68];
    __shared__ float mrow[32], lrow[32], arow[32];

    const int t = threadIdx.x, wave = t >> 5, lane = t & 31;
    const int bid = blockIdx.x;
    const int qt = bid & 31;                 // q-tile index == hq
    const int h  = (bid >> 5) % NH;
    const int b  = bid / (32 * NH);
    const long bh = (long)b * NH + h;
    const _Float16* qbh = q_ws + (bh * 1024 + qt * 32) * 64;  // [32][64]
    const _Float16* kbh = k_ws + bh * 1024 * 64;               // [1024][64]
    const _Float16* vbh = vt_ws + bh * 64 * 1024;              // [64][1024]

    // ---- Phase 0a: stage rel_pos_h/w as f16 (2 x 63 x 64 -> 2016 float4)
#pragma unroll
    for (int i = 0; i < 8; i++) {
        int idx = i * 256 + t;
        if (idx < 2016) {
            int tab = idx >= 1008;
            int rj  = tab ? idx - 1008 : idx;
            int r = rj >> 4, c4 = rj & 15;
            const float* src = (tab ? rel_pos_w : rel_pos_h) + r * 64 + c4 * 4;
            float4 v = *(const float4*)src;
            h4 hv = {(_Float16)v.x, (_Float16)v.y, (_Float16)v.z, (_Float16)v.w};
            *(h4*)&u.stage[tab][r][c4 * 4] = hv;
        }
    }
    if (t < 32) {                            // zero pad row 63 of both tables
        int tab = t >> 4, c4 = t & 15;
        h4 z = {(_Float16)0.f, (_Float16)0.f, (_Float16)0.f, (_Float16)0.f};
        *(h4*)&u.stage[tab][63][c4 * 4] = z;
    }
    if (t < 32) { mrow[t] = -3.0e38f; lrow[t] = 0.f; arow[t] = 0.f; }

    // Persistent Q fragments: 2 M-tiles x 2 k-steps (scaled q)
    h16 qa[2][2];
#pragma unroll
    for (int mt = 0; mt < 2; mt++)
#pragma unroll
        for (int ks = 0; ks < 2; ks++)
            qa[mt][ks] = load_a_frag(qbh + mt * 16 * 64 + ks * 32, 64, lane);

    f8 o;
#pragma unroll
    for (int e = 0; e < 8; e++) o[e] = 0.f;
    const int mt_o = wave >> 2, nt_o = wave & 3;   // O frag ownership
    __syncthreads();

    // ---- Phase 0b: rel bias GEMMs via WMMA (wave -> table = w>>2, j-tile = w&3)
    {
        const int tab = wave >> 2, ntj = wave & 3;
        f8 c0, c1;
#pragma unroll
        for (int e = 0; e < 8; e++) { c0[e] = 0.f; c1[e] = 0.f; }
#pragma unroll
        for (int ks = 0; ks < 2; ks++) {
            h16 bf = load_b_frag(&u.stage[tab][ntj * 16][0], 72, ks * 32, lane);
            c0 = wmma_f16(qa[0][ks], bf, c0);
            c1 = wmma_f16(qa[1][ks], bf, c1);
        }
        float (*relF)[68] = tab ? relWf : relHf;
        int nn = lane & 15, ho = (lane >> 4) << 3;
#pragma unroll
        for (int r = 0; r < 8; r++) {
            relF[r + ho][ntj * 16 + nn]      = c0[r];
            relF[16 + r + ho][ntj * 16 + nn] = c1[r];
        }
    }
    __syncthreads();   // stage reads done; S buffer (union) free for chunk loop

    for (int ch = 0; ch < 8; ch++) {
        const int kbase = ch * 128;
        // ---- S = Q K^T : wave owns key-column tile `wave`
        f8 c0, c1;
#pragma unroll
        for (int e = 0; e < 8; e++) { c0[e] = 0.f; c1[e] = 0.f; }
#pragma unroll
        for (int ks = 0; ks < 2; ks++) {
            h16 bfrag = load_b_frag(kbh + (kbase + wave * 16) * 64, 64, ks * 32, lane);
            c0 = wmma_f16(qa[0][ks], bfrag, c0);
            c1 = wmma_f16(qa[1][ks], bfrag, c1);
        }
        {
            int nn = lane & 15, ho = (lane >> 4) << 3;
#pragma unroll
            for (int r = 0; r < 8; r++) {
                u.S[r + ho][wave * 16 + nn]      = c0[r];
                u.S[16 + r + ho][wave * 16 + nn] = c1[r];
            }
        }
        __syncthreads();                                   // S ready

        // ---- online softmax: 8 threads per row, 16 cols each
        const int row = t >> 3, seg = t & 7, colb = seg * 16;
        const int hk = (kbase + colb) >> 5;
        const float bias_h = relHf[row][qt + 31 - hk];
        const int wbase = row + 31 - ((seg & 1) << 4);
        const float4* srow = (const float4*)&u.S[row][colb];
        float vreg[16], rmax = -3.0e38f;
#pragma unroll
        for (int j4 = 0; j4 < 4; j4++) {
            float4 sv = srow[j4];
            float sj[4] = {sv.x, sv.y, sv.z, sv.w};
#pragma unroll
            for (int q4 = 0; q4 < 4; q4++) {
                int j = j4 * 4 + q4;
                float val = sj[q4] + bias_h + relWf[row][wbase - j];
                vreg[j] = val;
                rmax = fmaxf(rmax, val);
            }
        }
        rmax = fmaxf(rmax, __shfl_xor(rmax, 1, 8));
        rmax = fmaxf(rmax, __shfl_xor(rmax, 2, 8));
        rmax = fmaxf(rmax, __shfl_xor(rmax, 4, 8));
        if (seg == 0) {
            float mo = mrow[row], mn = fmaxf(mo, rmax);
            mrow[row] = mn;
            arow[row] = __expf(mo - mn);
        }
        __syncthreads();                                   // stats ready

        const float mn = mrow[row];
        float psum = 0.f;
        h16 pvec;
#pragma unroll
        for (int j = 0; j < 16; j++) {
            float p = __expf(vreg[j] - mn);
            pvec[j] = (_Float16)p;
            psum += p;
        }
        *(h8*)&P[row][colb]     = *(h8*)&pvec;
        *(h8*)&P[row][colb + 8] = *((h8*)&pvec + 1);
        psum += __shfl_xor(psum, 1, 8);
        psum += __shfl_xor(psum, 2, 8);
        psum += __shfl_xor(psum, 4, 8);
        if (seg == 0) lrow[row] = lrow[row] * arow[row] + psum;

        {   // rescale O fragment by alpha
            int ho = (lane >> 4) << 3;
#pragma unroll
            for (int r = 0; r < 8; r++) o[r] *= arow[mt_o * 16 + r + ho];
        }
        __syncthreads();                                   // P ready

        // ---- O += P V : wave owns frag (mt_o, nt_o), 4 k-steps of 32 keys
#pragma unroll
        for (int ks = 0; ks < 4; ks++) {
            h16 pa = load_a_frag(&P[mt_o * 16][0] + ks * 32, 136, lane);
            h16 vb = load_b_frag(vbh + nt_o * 16 * 1024 + kbase, 1024, ks * 32, lane);
            o = wmma_f16(pa, vb, o);
        }
    }

    // ---- normalize + store [B,N,C]
    {
        int nn = lane & 15, ho = (lane >> 4) << 3;
#pragma unroll
        for (int r = 0; r < 8; r++) {
            int qrow = mt_o * 16 + r + ho;
            float val = o[r] / lrow[qrow];
            int n = qt * 32 + qrow;
            attn_out[((long)b * 1024 + n) * 768 + h * 64 + nt_o * 16 + nn] = val;
        }
    }
}

// ---------------------------------------------------------------------------
// Kernel 3: output projection  out = A[8192x768] * proj_w^T + b   (f32 out)
// ---------------------------------------------------------------------------
__global__ __launch_bounds__(256) void proj_kernel(
    const float* __restrict__ a, const float* __restrict__ w,
    const float* __restrict__ bias, float* __restrict__ out) {
    __shared__ _Float16 As[128][40];
    __shared__ _Float16 Ws[64][40];

    const int t = threadIdx.x, wave = t >> 5, lane = t & 31;
    const int row0 = blockIdx.x * 128;
    const int col0 = blockIdx.y * 64;

    f8 acc[4];
#pragma unroll
    for (int i = 0; i < 4; i++)
#pragma unroll
        for (int e = 0; e < 8; e++) acc[i][e] = 0.f;

    for (int kb = 0; kb < 768; kb += 32) {
        float4 xv[4];
#pragma unroll
        for (int i = 0; i < 4; i++) {
            int idx = i * 256 + t, r = idx >> 3, c4 = idx & 7;
            xv[i] = *(const float4*)&a[(long)(row0 + r) * 768 + kb + c4 * 4];
        }
        float4 wv[2];
#pragma unroll
        for (int i = 0; i < 2; i++) {
            int idx = i * 256 + t, r = idx >> 3, c4 = idx & 7;
            wv[i] = *(const float4*)&w[(long)(col0 + r) * 768 + kb + c4 * 4];
        }
        if (kb + 32 < 768)
            __builtin_prefetch(&a[(long)(row0 + (t >> 1)) * 768 + kb + 32], 0, 1);
#pragma unroll
        for (int i = 0; i < 4; i++) {
            int idx = i * 256 + t, r = idx >> 3, c4 = idx & 7;
            h4 hv = {(_Float16)xv[i].x, (_Float16)xv[i].y,
                     (_Float16)xv[i].z, (_Float16)xv[i].w};
            *(h4*)&As[r][c4 * 4] = hv;
        }
#pragma unroll
        for (int i = 0; i < 2; i++) {
            int idx = i * 256 + t, r = idx >> 3, c4 = idx & 7;
            h4 hv = {(_Float16)wv[i].x, (_Float16)wv[i].y,
                     (_Float16)wv[i].z, (_Float16)wv[i].w};
            *(h4*)&Ws[r][c4 * 4] = hv;
        }
        __syncthreads();
        h16 afrag = load_a_frag(&As[wave * 16][0], 40, lane);
#pragma unroll
        for (int nt = 0; nt < 4; nt++) {
            h16 bfrag = load_b_frag(&Ws[nt * 16][0], 40, 0, lane);
            acc[nt] = wmma_f16(afrag, bfrag, acc[nt]);
        }
        __syncthreads();
    }

    const int nn = lane & 15;
    const int ho = (lane >> 4) << 3;
#pragma unroll
    for (int nt = 0; nt < 4; nt++) {
        int cout = col0 + nt * 16 + nn;
        float bv = bias[cout];
#pragma unroll
        for (int r = 0; r < 8; r++) {
            int m = row0 + wave * 16 + r + ho;
            out[(long)m * 768 + cout] = acc[nt][r] + bv;
        }
    }
}

// ---------------------------------------------------------------------------
extern "C" void kernel_launch(void* const* d_in, const int* in_sizes, int n_in,
                              void* d_out, int out_size, void* d_ws, size_t ws_size,
                              hipStream_t stream) {
    (void)in_sizes; (void)n_in; (void)out_size; (void)ws_size;
    const float* x         = (const float*)d_in[0];
    const float* qkv_w     = (const float*)d_in[1];
    const float* rel_pos_h = (const float*)d_in[2];
    const float* rel_pos_w = (const float*)d_in[3];
    const float* proj_w    = (const float*)d_in[4];
    const float* proj_b    = (const float*)d_in[5];

    char* ws = (char*)d_ws;
    const size_t QKV_BYTES = (size_t)8 * NH * 1024 * HD * sizeof(_Float16); // 12.58 MB each
    _Float16* q_ws   = (_Float16*)(ws);
    _Float16* k_ws   = (_Float16*)(ws + QKV_BYTES);
    _Float16* vt_ws  = (_Float16*)(ws + 2 * QKV_BYTES);
    float*    attn_o = (float*)   (ws + 3 * QKV_BYTES);
    float*    out    = (float*)d_out;

    dim3 g1(64, 36);   // 8192/128 x 2304/64
    qkv_kernel<<<g1, 256, 0, stream>>>(x, qkv_w, q_ws, k_ws, vt_ws);

    attn_kernel<<<8 * NH * 32, 256, 0, stream>>>(q_ws, k_ws, vt_ws,
                                                 rel_pos_h, rel_pos_w, attn_o);

    dim3 g3(64, 12);   // 8192/128 x 768/64
    proj_kernel<<<g3, 256, 0, stream>>>(attn_o, proj_w, proj_b, out);
}
